// CA_Model_15899968930269
// MI455X (gfx1250) — matrix operand order repaired and checked by reading, not compiled, not run on hardware
//
#include <hip/hip_runtime.h>
#include <hip/hip_bf16.h>
#include <math.h>

// ---------------------------------------------------------------------------
// Neural CA on MI455X (gfx1250): bf16 WMMA for the 95->128->16 MLP,
// async-to-LDS staged perception conv, 2 kernels per step, hidden ping-pong.
// ---------------------------------------------------------------------------

#define NC       3
#define HIDDEN   16
#define STATE    19     // NC + HIDDEN
#define PV       5
#define PERC     95
#define PERC_PAD 96     // pad K to 3 x 32 for wmma 16x16x32
#define MID      128
#define IMG      128
#define BATCH    4
#define STEPS    16

// tile: 16 wide x 8 high = 128 pixels; 8 waves (wave32) -> 1 M-tile per wave
#define TW 16
#define TH 8
#define NTHREADS 256

typedef __attribute__((ext_vector_type(16))) __bf16 v16bf;
typedef __attribute__((ext_vector_type(8)))  float  v8f;

// ---------------------------------------------------------------------------
// Optional CDNA5 async global->LDS copies (ASYNCcnt path). Guarded so the
// kernel still compiles (falling back to ld+ds_store) if the toolchain lacks
// the builtins. Probe result: builtin exists and takes (global int*, lds int*,
// imm offset, imm cpol).
// ---------------------------------------------------------------------------
#ifndef __has_builtin
#define __has_builtin(x) 0
#endif
#if __has_builtin(__builtin_amdgcn_global_load_async_to_lds_b32) && \
    __has_builtin(__builtin_amdgcn_s_wait_asynccnt)
#define HAVE_ASYNC_LDS 1
#else
#define HAVE_ASYNC_LDS 0
#endif

typedef __attribute__((address_space(1))) int* ca_gptr_i32;
typedef __attribute__((address_space(3))) int* ca_lptr_i32;

__device__ __forceinline__ void ca_copy_f32_to_lds(float* lds_dst,
                                                   const float* gsrc) {
#if HAVE_ASYNC_LDS
  // global_load_async_to_lds_b32: (global src, lds dst, imm offset, imm cpol)
  __builtin_amdgcn_global_load_async_to_lds_b32(
      (ca_gptr_i32)gsrc, (ca_lptr_i32)lds_dst, 0, 0);
#else
  *lds_dst = *gsrc;
#endif
}

__device__ __forceinline__ void ca_async_wait_all() {
#if HAVE_ASYNC_LDS
  __builtin_amdgcn_s_wait_asynccnt(0);
#endif
}

// Shared memory layout (57344 B total):
//   [0,      24576) : Abuf  : bf16 [8 mtile][3 kblk][32 lane][16 half] (A frags)
//   [24576,  57344) : phase1: Sbuf f32[19][10][18] + Wp f32[855] + bp f32[95]
//                     phase2: Hbuf bf16 [8 wave][4 kblk][32 lane][16 half]
#define SMEM_BYTES 57344
#define OFF_S      24576
#define OFF_WP     (OFF_S + 19 * 10 * 18 * 4)   // 38256
#define OFF_BP     (OFF_WP + PERC * 9 * 4)      // 41676

// ---------------------------------------------------------------------------
// Pre-pack W1 / W2 into WMMA B-fragment layout (bf16), once per launch.
//   B frag (16-bit, 32x16): lane half g=lane>>4, half h -> K=8g+(h&7)+16*(h>>3);
//   N = lane & 15.
// ---------------------------------------------------------------------------
__global__ __launch_bounds__(NTHREADS) void ca_prep_weights(
    const float* __restrict__ W1, const float* __restrict__ W2,
    __bf16* __restrict__ B1f, __bf16* __restrict__ B2f) {
  int tid = blockIdx.x * NTHREADS + threadIdx.x;
  int stride = gridDim.x * NTHREADS;
  // W1: (128,95) -> 8 ntiles x 3 kblks x 512 halves
  for (int i = tid; i < 8 * 3 * 512; i += stride) {
    int nt = i / (3 * 512);
    int r  = i - nt * (3 * 512);
    int kb = r / 512;
    int li = r - kb * 512;
    int lane = li >> 4, h = li & 15;
    int n = nt * 16 + (lane & 15);
    int k = kb * 32 + ((lane >> 4) << 3) + (h & 7) + ((h >> 3) << 4);
    float v = (k < PERC) ? W1[n * PERC + k] : 0.0f;
    B1f[i] = (__bf16)v;
  }
  // W2: (16,128) -> 4 kblks x 512 halves
  for (int i = tid; i < 4 * 512; i += stride) {
    int kb = i / 512;
    int li = i - kb * 512;
    int lane = li >> 4, h = li & 15;
    int n = lane & 15;
    int k = kb * 32 + ((lane >> 4) << 3) + (h & 7) + ((h >> 3) << 4);
    B2f[i] = (__bf16)W2[n * MID + k];
  }
}

__global__ __launch_bounds__(NTHREADS) void ca_zero(float* __restrict__ p, int n) {
  int i = blockIdx.x * NTHREADS + threadIdx.x;
  if (i < n) p[i] = 0.0f;
}

// ---------------------------------------------------------------------------
// Step kernel 1: perception conv + MLP via WMMA, hidden += update (pre-mask).
// Grid: (IMG/TW=8, IMG/TH=16, BATCH)
// ---------------------------------------------------------------------------
__global__ __launch_bounds__(NTHREADS) void ca_step_update(
    const float* __restrict__ xin,   // (4,3,128,128) color, immutable
    const float* __restrict__ Wp,    // (95,1,3,3)
    const float* __restrict__ bp,    // (95,)
    const float* __restrict__ bias1, // (128,)
    const float* __restrict__ bias2, // (16,)
    const __bf16* __restrict__ B1f,  // W1 B-fragments
    const __bf16* __restrict__ B2f,  // W2 B-fragments
    const float* __restrict__ hin,   // (4,16,128,128) hidden in
    float* __restrict__ hmid)        // (4,16,128,128) hidden+upd (pre-mask)
{
  __shared__ unsigned char smem[SMEM_BYTES] __attribute__((aligned(32)));
  __bf16* Abuf = (__bf16*)smem;
  float*  Sbuf = (float*)(smem + OFF_S);
  float*  Wpb  = (float*)(smem + OFF_WP);
  float*  bpb  = (float*)(smem + OFF_BP);

  const int tid  = threadIdx.x;
  const int lane = tid & 31;
  const int wv   = tid >> 5;
  const int tx0  = blockIdx.x * TW;
  const int ty0  = blockIdx.y * TH;
  const int b    = blockIdx.z;

  __builtin_prefetch(B1f, 0, 3);  // global_prefetch_b8 -> warm GL2 for weights

  // ---- phase 0: stage state tile (+1 halo) and conv weights into LDS ------
  // In-bounds elements go through the async global->LDS DMA (ASYNCcnt);
  // out-of-bounds halo entries are zeroed with a normal DS store. Each LDS
  // slot is written by exactly one mechanism, so no DS/ASYNC ordering hazard.
  for (int i = tid; i < STATE * 10 * 18; i += NTHREADS) {
    int ch = i / 180;
    int r  = i - ch * 180;
    int sy = r / 18;
    int sx = r - sy * 18;
    int gy = ty0 + sy - 1, gx = tx0 + sx - 1;
    if (gy >= 0 && gy < IMG && gx >= 0 && gx < IMG) {
      const float* src = (ch < NC)
          ? &xin[((b * NC + ch) * IMG + gy) * IMG + gx]
          : &hin[((b * HIDDEN + (ch - NC)) * IMG + gy) * IMG + gx];
      ca_copy_f32_to_lds(&Sbuf[i], src);
    } else {
      Sbuf[i] = 0.0f;
    }
  }
  for (int i = tid; i < PERC * 9; i += NTHREADS) ca_copy_f32_to_lds(&Wpb[i], &Wp[i]);
  for (int i = tid; i < PERC; i += NTHREADS)     ca_copy_f32_to_lds(&bpb[i], &bp[i]);
  ca_async_wait_all();   // s_wait_asynccnt 0
  __syncthreads();

  // ---- phase 1: grouped 3x3 conv -> Abuf in WMMA A-fragment bf16 layout ----
  // A frag (16-bit, 16x32): lane = M + (k%32 has bit3 ? 16 : 0),
  //                         half = (k&7) + ((k&16)>>1)
  for (int j = tid; j < TW * TH * PERC_PAD; j += NTHREADS) {
    int pix = j / PERC_PAD;
    int o   = j - pix * PERC_PAD;
    int m = pix & 15;      // lx, also M row
    int w = pix >> 4;      // ly, also mtile id
    float v = 0.0f;
    if (o < PERC) {
      int ci = o / PV;
      const float* wp9 = &Wpb[o * 9];
      const float* sb  = &Sbuf[(ci * 10 + w) * 18 + m];
      float acc = bpb[o];
#pragma unroll
      for (int ky = 0; ky < 3; ++ky)
#pragma unroll
        for (int kx = 0; kx < 3; ++kx)
          acc = fmaf(sb[ky * 18 + kx], wp9[ky * 3 + kx], acc);
      v = acc;
    }
    int kb = o >> 5, kk = o & 31;
    int hh = (kk & 7) + ((kk & 16) >> 1);
    int ln = m + ((kk & 8) ? 16 : 0);
    Abuf[((w * 3 + kb) * 32 + ln) * 16 + hh] = (__bf16)v;
  }
  __syncthreads();

  // ---- phase 2: GEMM1  H(16x128) = P(16x96) x W1^T, per wave ----
  const v16bf a0 = *(const v16bf*)&Abuf[((wv * 3 + 0) * 32 + lane) * 16];
  const v16bf a1 = *(const v16bf*)&Abuf[((wv * 3 + 1) * 32 + lane) * 16];
  const v16bf a2 = *(const v16bf*)&Abuf[((wv * 3 + 2) * 32 + lane) * 16];

  __bf16* Hbuf = (__bf16*)(smem + OFF_S);  // reuse Sbuf/Wp region (post-barrier)
  const int n_lo = lane & 15;
  const int mrow = (lane >> 4) * 8;

#pragma unroll
  for (int nt = 0; nt < 8; ++nt) {
    v8f acc = {};
    const v16bf bA = *(const v16bf*)&B1f[((nt * 3 + 0) * 32 + lane) * 16];
    const v16bf bB = *(const v16bf*)&B1f[((nt * 3 + 1) * 32 + lane) * 16];
    const v16bf bC = *(const v16bf*)&B1f[((nt * 3 + 2) * 32 + lane) * 16];
    acc = __builtin_amdgcn_wmma_f32_16x16x32_bf16(false, a0, false, bA, (short)0, acc, false, false);
    acc = __builtin_amdgcn_wmma_f32_16x16x32_bf16(false, a1, false, bB, (short)0, acc, false, false);
    acc = __builtin_amdgcn_wmma_f32_16x16x32_bf16(false, a2, false, bC, (short)0, acc, false, false);

    const float bv = bias1[nt * 16 + n_lo];
    // restage H column (channel kg) into A-fragment layout in LDS
    const int kg  = nt * 16 + n_lo;
    const int kb2 = kg >> 5, kk = kg & 31;
    const int hh  = (kk & 7) + ((kk & 16) >> 1);
    const int lnb = (kk & 8) ? 16 : 0;
#pragma unroll
    for (int r = 0; r < 8; ++r) {
      float h = acc[r] + bv;
      h = (h >= 0.0f) ? h : 0.2f * h;           // LeakyReLU(0.2)
      int Mm = mrow + r;
      Hbuf[((wv * 4 + kb2) * 32 + (Mm + lnb)) * 16 + hh] = (__bf16)h;
    }
  }
  __syncthreads();

  // ---- phase 3: GEMM2  U(16x16) = H(16x128) x W2^T, then hidden += U ----
  v8f acc2 = {};
#pragma unroll
  for (int kb2 = 0; kb2 < 4; ++kb2) {
    const v16bf ah = *(const v16bf*)&Hbuf[((wv * 4 + kb2) * 32 + lane) * 16];
    const v16bf bw = *(const v16bf*)&B2f[(kb2 * 32 + lane) * 16];
    acc2 = __builtin_amdgcn_wmma_f32_16x16x32_bf16(false, ah, false, bw, (short)0, acc2, false, false);
  }
  const float bv2 = bias2[n_lo];
  const int gy = ty0 + wv;
#pragma unroll
  for (int r = 0; r < 8; ++r) {
    int Mm = mrow + r;
    int gx = tx0 + Mm;
    int idx = ((b * HIDDEN + n_lo) * IMG + gy) * IMG + gx;
    hmid[idx] = hin[idx] + acc2[r] + bv2;
  }
}

// ---------------------------------------------------------------------------
// Step kernel 2: alive mask (3x3 maxpool of hidden ch1 > 0) + sigmoid on ch1.
// ---------------------------------------------------------------------------
__global__ __launch_bounds__(NTHREADS) void ca_step_mask(
    const float* __restrict__ hmid, float* __restrict__ hout) {
  int i = blockIdx.x * NTHREADS + threadIdx.x;
  if (i >= BATCH * IMG * IMG) return;
  int x = i & (IMG - 1);
  int y = (i >> 7) & (IMG - 1);
  int b = i >> 14;
  const float* ch1 = hmid + (b * HIDDEN + 1) * IMG * IMG;
  float mx = -3.4028235e38f;
#pragma unroll
  for (int dy = -1; dy <= 1; ++dy)
#pragma unroll
    for (int dx = -1; dx <= 1; ++dx) {
      int yy = y + dy, xx = x + dx;
      if (yy >= 0 && yy < IMG && xx >= 0 && xx < IMG) {
        float v = ch1[yy * IMG + xx];
        mx = (v > mx) ? v : mx;
      }
    }
  float alive = (mx > 0.0f) ? 1.0f : 0.0f;
#pragma unroll
  for (int c = 0; c < HIDDEN; ++c) {
    int idx = ((b * HIDDEN + c) * IMG + y) * IMG + x;
    float v = hmid[idx] * alive;
    if (c == 1) v = 1.0f / (1.0f + __expf(-v));  // sigmoid on state ch 4
    hout[idx] = v;
  }
}

// ---------------------------------------------------------------------------
// Final pack: out = concat(colors x, hidden) -> (4,19,128,128)
// ---------------------------------------------------------------------------
__global__ __launch_bounds__(NTHREADS) void ca_pack(
    const float* __restrict__ xin, const float* __restrict__ h,
    float* __restrict__ out) {
  int i = blockIdx.x * NTHREADS + threadIdx.x;
  if (i >= BATCH * STATE * IMG * IMG) return;
  int sp = i % (IMG * IMG);
  int c  = (i / (IMG * IMG)) % STATE;
  int b  = i / (IMG * IMG * STATE);
  out[i] = (c < NC) ? xin[(b * NC + c) * IMG * IMG + sp]
                    : h[(b * HIDDEN + (c - NC)) * IMG * IMG + sp];
}

// ---------------------------------------------------------------------------
extern "C" void kernel_launch(void* const* d_in, const int* in_sizes, int n_in,
                              void* d_out, int out_size, void* d_ws, size_t ws_size,
                              hipStream_t stream) {
  const float* x   = (const float*)d_in[0];
  const float* Wp  = (const float*)d_in[1];
  const float* bp  = (const float*)d_in[2];
  const float* W1  = (const float*)d_in[3];
  const float* b1  = (const float*)d_in[4];
  const float* W2  = (const float*)d_in[5];
  const float* b2  = (const float*)d_in[6];
  // d_in[7] = steps (always 16 in the reference setup; graph-captured)

  char* ws = (char*)d_ws;
  __bf16* B1f = (__bf16*)ws;                       // 24576 B
  __bf16* B2f = (__bf16*)(ws + 24576);             //  4096 B
  float*  hA  = (float*)(ws + 32768);              // 4 MiB hidden state
  float*  hM  = (float*)(ws + 32768 + (size_t)BATCH * HIDDEN * IMG * IMG * 4);

  const int hidden_elems = BATCH * HIDDEN * IMG * IMG;  // 1,048,576

  ca_prep_weights<<<48, NTHREADS, 0, stream>>>(W1, W2, B1f, B2f);
  ca_zero<<<(hidden_elems + NTHREADS - 1) / NTHREADS, NTHREADS, 0, stream>>>(hA, hidden_elems);

  dim3 grid_upd(IMG / TW, IMG / TH, BATCH);  // (8,16,4)
  const int npix = BATCH * IMG * IMG;        // 65,536
  for (int s = 0; s < STEPS; ++s) {
    ca_step_update<<<grid_upd, NTHREADS, 0, stream>>>(x, Wp, bp, b1, b2, B1f, B2f, hA, hM);
    ca_step_mask<<<(npix + NTHREADS - 1) / NTHREADS, NTHREADS, 0, stream>>>(hM, hA);
  }

  const int nout = BATCH * STATE * IMG * IMG;
  ca_pack<<<(nout + NTHREADS - 1) / NTHREADS, NTHREADS, 0, stream>>>(x, hA, (float*)d_out);
}
